// LSTM_Model_24489903522697
// MI455X (gfx1250) — matrix operand order repaired and checked by reading, not compiled, not run on hardware
//
#include <hip/hip_runtime.h>
#include <hip/hip_bf16.h>

// ---------------- problem constants ----------------
#define BATCH   64
#define SEQLEN  128
#define EMBED   300
#define EPAD    320            // embed K padded to multiple of 32 (zero fill)
#define HID     512
#define K0PAD   832            // EPAD + HID
#define K1PAD   1024           // HID + HID
#define ZCOLS   2048           // 4 gates * HID
#define NWG     32             // persistent workgroups (co-resident)
#define COLSWG  16             // h-columns owned per WG  (NWG*COLSWG == HID)
#define NTHREADS 256           // 8 waves (wave32)
#define K0TILES 26             // 832/32
#define K1TILES 32             // 1024/32
#define XE_KT   10             // EPAD/32

typedef __bf16 bf16_t;
typedef bf16_t v16bf __attribute__((ext_vector_type(16)));
typedef float  v8f   __attribute__((ext_vector_type(8)));

union V8 { v8f v; float f[8]; };

// ---------------- WMMA helpers ----------------
__device__ inline v8f wmma_bf16(v16bf a, v16bf b, v8f c) {
  // D = A(16x32 bf16) * B(32x16 bf16) + C(16x16 f32)
  return __builtin_amdgcn_wmma_f32_16x16x32_bf16(
      /*neg_a=*/false, a, /*neg_b=*/false, b,
      /*c_mod=*/(short)0, c, /*reuse_a=*/false, /*reuse_b=*/false);
}

// A fragment (16x32 bf16, row-major source, ld in elements).
// Layout: lanes 0-15 hold K 0..15, lanes 16-31 hold K 16..31 (matches load_b_frag).
__device__ inline v16bf load_a_frag(const bf16_t* base, int ld, int m0, int k0, int lane) {
  const int m   = m0 + (lane & 15);
  const int hlf = lane >> 4;
  const bf16_t* row = base + (size_t)m * ld + k0 + hlf * 16;
  union { uint4 u[2]; v16bf v; } r;
  r.u[0] = *(const uint4*)(row);        // elements 0..7
  r.u[1] = *(const uint4*)(row + 8);    // elements 8..15
  return r.v;
}

// B fragment (32x16 bf16) from K-contiguous (column-major-of-original) weights:
// wt[n*ldk + k].  lanes 0-15: K 0..15, lanes 16-31: K 16..31, N = lane&15.
__device__ inline v16bf load_b_frag(const bf16_t* wt, int ldk, int n0, int k0, int lane) {
  const bf16_t* p = wt + (size_t)(n0 + (lane & 15)) * ldk + k0 + (lane >> 4) * 16;
  union { uint4 u[2]; v16bf v; } r;
  r.u[0] = ((const uint4*)p)[0];
  r.u[1] = ((const uint4*)p)[1];
  return r.v;
}

__device__ inline float sigm(float x) { return 1.0f / (1.0f + __expf(-x)); }

// Device-wide barrier: 32 co-resident WGs, monotonically increasing target.
__device__ inline void grid_barrier(unsigned* cnt, unsigned target) {
  __threadfence();                 // release stores (h writes) to device scope
  __syncthreads();
  if (threadIdx.x == 0) {
    atomicAdd(cnt, 1u);
    while (atomicAdd(cnt, 0u) < target) __builtin_amdgcn_s_sleep(2);
    __threadfence();               // acquire: invalidate stale WGP$ lines
  }
  __syncthreads();
}

// ---------------- prologue kernels ----------------
__global__ void zero_words(unsigned* p, int nwords) {
  for (int i = blockIdx.x * blockDim.x + threadIdx.x; i < nwords;
       i += gridDim.x * blockDim.x) p[i] = 0u;
}

// W0 [812 x 2048] row-major f32 -> Wt0 [2048][832] bf16, K padded (300->320 zero fill)
__global__ void pack_w0(const float* __restrict__ W0, bf16_t* __restrict__ Wt0) {
  const int total = ZCOLS * K0PAD;
  for (int i = blockIdx.x * blockDim.x + threadIdx.x; i < total;
       i += gridDim.x * blockDim.x) {
    const int n = i / K0PAD, k = i - n * K0PAD;
    float v = 0.f;
    if (k < EPAD) { if (k < EMBED) v = W0[(size_t)k * ZCOLS + n]; }
    else           { v = W0[(size_t)(EMBED + (k - EPAD)) * ZCOLS + n]; }
    Wt0[i] = (bf16_t)v;
  }
}

// W1 [1024 x 2048] row-major f32 -> Wt1 [2048][1024] bf16
__global__ void pack_w1(const float* __restrict__ W1, bf16_t* __restrict__ Wt1) {
  const int total = ZCOLS * K1PAD;
  for (int i = blockIdx.x * blockDim.x + threadIdx.x; i < total;
       i += gridDim.x * blockDim.x) {
    const int n = i / K1PAD, k = i - n * K1PAD;
    Wt1[i] = (bf16_t)W1[(size_t)k * ZCOLS + n];
  }
}

// xe[t][b][k] bf16, k padded to EPAD with zeros
__global__ void embed_gather(const int* __restrict__ x, const float* __restrict__ emb,
                             bf16_t* __restrict__ xe) {
  const int total = SEQLEN * BATCH * EPAD;
  for (int i = blockIdx.x * blockDim.x + threadIdx.x; i < total;
       i += gridDim.x * blockDim.x) {
    const int k   = i % EPAD;
    const int tb  = i / EPAD;
    const int b   = tb % BATCH;
    const int t   = tb / BATCH;
    float v = 0.f;
    if (k < EMBED) {
      const int idx = x[b * SEQLEN + t];          // x is [BATCH, SEQ]
      v = emb[(size_t)idx * EMBED + k];
    }
    xe[i] = (bf16_t)v;
  }
}

// ---------------- persistent LSTM kernel ----------------
__global__ __launch_bounds__(NTHREADS, 1)
void lstm_persistent(const bf16_t* __restrict__ xe,
                     const bf16_t* __restrict__ Wt0,
                     const bf16_t* __restrict__ Wt1,
                     bf16_t* __restrict__ h0buf,    // 2 ping-pong buffers [64][512]
                     bf16_t* __restrict__ h1buf,    // 2 ping-pong buffers [64][512]
                     const float* __restrict__ b0,
                     const float* __restrict__ b1,
                     unsigned* __restrict__ barcnt)
{
  __shared__ float zls[4 * 4 * 256];     // [gate][mtile][m][n]  16 KB

  const int tid  = threadIdx.x;
  const int wave = tid >> 5;
  const int lane = tid & 31;
  const int hlf  = lane >> 4;
  const int nlo  = lane & 15;
  const int mt   = wave & 3;             // M tile this wave computes (GEMM phase)
  const int g0   = (wave >> 2) * 2;      // gate pair {0,1} or {2,3}
  const int colbase = blockIdx.x * COLSWG;

  // per-lane gate biases for the column this lane updates (n = colbase+nlo)
  float bia[4], bib[4];
  for (int g = 0; g < 4; ++g) {
    bia[g] = b0[g * HID + colbase + nlo];
    bib[g] = b1[g * HID + colbase + nlo];
  }

  // cell state lives in registers as C-fragments for the whole sequence
  V8 c0, c1;
  for (int r = 0; r < 8; ++r) { c0.f[r] = 0.f; c1.f[r] = 0.f; }

  unsigned target = 0;

  for (int t = 0; t < SEQLEN; ++t) {
    const int p = t & 1;
    const bf16_t* h0prev = h0buf + (size_t)p       * (BATCH * HID);
    bf16_t*       h0cur  = h0buf + (size_t)(1 - p) * (BATCH * HID);
    const bf16_t* h1prev = h1buf + (size_t)p       * (BATCH * HID);
    bf16_t*       h1cur  = h1buf + (size_t)(1 - p) * (BATCH * HID);
    const bf16_t* xet    = xe + (size_t)t * (BATCH * EPAD);

    // Launder a zero element-offset through inline asm once per timestep.
    // The opaque, per-iteration value defeats LICM of the weight B-fragments
    // across the 128-step loop (which previously spilled ~3.7KB/lane to
    // scratch), while *keeping pointer provenance* so infer-address-spaces
    // still emits global_load_b128 / global_prefetch (not flat_*, which
    // would couple the weight stream to DScnt and the LDS pipe).
    unsigned long long zofs = 0;
    asm volatile("" : "+s"(zofs) :: "memory");
    const bf16_t* wt0p = Wt0 + zofs;
    const bf16_t* wt1p = Wt1 + zofs;

    // ======== layer 0: z0 = [x_t | h0prev] @ W0 (this WG's 4x16 gate cols) ========
    {
      v8f acc0 = {}; v8f acc1 = {};
      for (int kt = 0; kt < K0TILES; ++kt) {
        const int kk = kt * 32;
        v16bf af = (kt < XE_KT)
            ? load_a_frag(xet,    EPAD, mt * 16, kk, lane)
            : load_a_frag(h0prev, HID,  mt * 16, (kt - XE_KT) * 32, lane);
        v16bf bA = load_b_frag(wt0p, K0PAD, (g0    ) * HID + colbase, kk, lane);
        v16bf bB = load_b_frag(wt0p, K0PAD, (g0 + 1) * HID + colbase, kk, lane);
        if (kt + 1 < K0TILES)   // gfx1250 global_prefetch on the weight stream
          __builtin_prefetch(wt0p + (size_t)(g0 * HID + colbase + nlo) * K0PAD + kk + 32, 0, 3);
        acc0 = wmma_bf16(af, bA, acc0);
        acc1 = wmma_bf16(af, bB, acc1);
      }
      V8 s0, s1; s0.v = acc0; s1.v = acc1;
      for (int r = 0; r < 8; ++r) {
        const int m = r + hlf * 8;
        zls[((g0    ) * 4 + mt) * 256 + m * 16 + nlo] = s0.f[r];
        zls[((g0 + 1) * 4 + mt) * 256 + m * 16 + nlo] = s1.f[r];
      }
    }
    __syncthreads();

    // layer-0 cell update: waves 0..3, one M tile each, fused in fragment layout
    if (wave < 4) {
      const int umt = wave;
      for (int r = 0; r < 8; ++r) {
        const int m  = r + hlf * 8;
        const int zi = umt * 256 + m * 16 + nlo;        // gate stride = 1024 floats
        const float gi = zls[zi         ] + bia[0];
        const float gj = zls[zi + 1024  ] + bia[1];
        const float gf = zls[zi + 2048  ] + bia[2];
        const float go = zls[zi + 3072  ] + bia[3];
        const float c  = c0.f[r] * sigm(gf + 1.0f) + sigm(gi) * tanhf(gj);
        c0.f[r] = c;
        const float h = tanhf(c) * sigm(go);
        h0cur[(size_t)(umt * 16 + m) * HID + colbase + nlo] = (bf16_t)h;
      }
    }
    target += NWG; grid_barrier(barcnt, target);

    // ======== layer 1: z1 = [h0cur | h1prev] @ W1 ========
    {
      v8f acc0 = {}; v8f acc1 = {};
      for (int kt = 0; kt < K1TILES; ++kt) {
        const int kk = kt * 32;
        v16bf af = (kt < 16)
            ? load_a_frag(h0cur,  HID, mt * 16, kk, lane)
            : load_a_frag(h1prev, HID, mt * 16, (kt - 16) * 32, lane);
        v16bf bA = load_b_frag(wt1p, K1PAD, (g0    ) * HID + colbase, kk, lane);
        v16bf bB = load_b_frag(wt1p, K1PAD, (g0 + 1) * HID + colbase, kk, lane);
        if (kt + 1 < K1TILES)
          __builtin_prefetch(wt1p + (size_t)(g0 * HID + colbase + nlo) * K1PAD + kk + 32, 0, 3);
        acc0 = wmma_bf16(af, bA, acc0);
        acc1 = wmma_bf16(af, bB, acc1);
      }
      V8 s0, s1; s0.v = acc0; s1.v = acc1;
      for (int r = 0; r < 8; ++r) {
        const int m = r + hlf * 8;
        zls[((g0    ) * 4 + mt) * 256 + m * 16 + nlo] = s0.f[r];
        zls[((g0 + 1) * 4 + mt) * 256 + m * 16 + nlo] = s1.f[r];
      }
    }
    __syncthreads();

    // layer-1 cell update: waves 4..7
    if (wave >= 4) {
      const int umt = wave - 4;
      for (int r = 0; r < 8; ++r) {
        const int m  = r + hlf * 8;
        const int zi = umt * 256 + m * 16 + nlo;
        const float gi = zls[zi         ] + bib[0];
        const float gj = zls[zi + 1024  ] + bib[1];
        const float gf = zls[zi + 2048  ] + bib[2];
        const float go = zls[zi + 3072  ] + bib[3];
        const float c  = c1.f[r] * sigm(gf + 1.0f) + sigm(gi) * tanhf(gj);
        c1.f[r] = c;
        const float h = tanhf(c) * sigm(go);
        h1cur[(size_t)(umt * 16 + m) * HID + colbase + nlo] = (bf16_t)h;
      }
    }
    target += NWG; grid_barrier(barcnt, target);
  }
}

// ---------------- tiny heads: feats = h1@Wd+bd -> two softmaxes ----------------
__global__ void head_kernel(const bf16_t* __restrict__ h1,   // final buffer (buf 0)
                            const float* __restrict__ Wd,  const float* __restrict__ bd,
                            const float* __restrict__ Wpe, const float* __restrict__ bpe,
                            const float* __restrict__ Wbur,const float* __restrict__ bbur,
                            float* __restrict__ out)
{
  const int b = threadIdx.x;
  if (b >= BATCH) return;
  float f0 = bd[0], f1 = bd[1];
  for (int k = 0; k < HID; ++k) {
    const float h = (float)h1[(size_t)b * HID + k];
    f0 += h * Wd[k * 2 + 0];
    f1 += h * Wd[k * 2 + 1];
  }
  // pred_pe (2-way softmax)
  const float l0 = f0 * Wpe[0] + f1 * Wpe[2] + bpe[0];
  const float l1 = f0 * Wpe[1] + f1 * Wpe[3] + bpe[1];
  const float mx = fmaxf(l0, l1);
  const float e0 = __expf(l0 - mx), e1 = __expf(l1 - mx);
  const float inv = 1.0f / (e0 + e1);
  out[b * 2 + 0] = e0 * inv;
  out[b * 2 + 1] = e1 * inv;
  // pred_burden (5-way softmax)
  float lb[5]; float mb = -1e30f;
  for (int j = 0; j < 5; ++j) {
    lb[j] = f0 * Wbur[j] + f1 * Wbur[5 + j] + bbur[j];
    mb = fmaxf(mb, lb[j]);
  }
  float s = 0.f;
  for (int j = 0; j < 5; ++j) { lb[j] = __expf(lb[j] - mb); s += lb[j]; }
  const float invs = 1.0f / s;
  float* outb = out + BATCH * 2;
  for (int j = 0; j < 5; ++j) outb[b * 5 + j] = lb[j] * invs;
}

// ---------------- host launcher ----------------
extern "C" void kernel_launch(void* const* d_in, const int* in_sizes, int n_in,
                              void* d_out, int out_size, void* d_ws, size_t ws_size,
                              hipStream_t stream) {
  (void)in_sizes; (void)n_in; (void)out_size; (void)ws_size;
  const int*   x    = (const int*)  d_in[0];
  const float* emb  = (const float*)d_in[1];
  const float* W0   = (const float*)d_in[2];
  const float* b0   = (const float*)d_in[3];
  const float* W1   = (const float*)d_in[4];
  const float* b1   = (const float*)d_in[5];
  const float* Wd   = (const float*)d_in[6];
  const float* bd   = (const float*)d_in[7];
  const float* Wpe  = (const float*)d_in[8];
  const float* bpe  = (const float*)d_in[9];
  const float* Wbur = (const float*)d_in[10];
  const float* bbur = (const float*)d_in[11];
  float* out = (float*)d_out;

  char* ws = (char*)d_ws;
  size_t off = 0;
  auto take = [&](size_t bytes) -> char* {
    char* p = ws + off;
    off = (off + bytes + 255) & ~(size_t)255;
    return p;
  };
  bf16_t*   xe  = (bf16_t*)take((size_t)SEQLEN * BATCH * EPAD * 2);  // 5.24 MB
  bf16_t*   Wt0 = (bf16_t*)take((size_t)ZCOLS * K0PAD * 2);          // 3.41 MB
  bf16_t*   Wt1 = (bf16_t*)take((size_t)ZCOLS * K1PAD * 2);          // 4.19 MB
  bf16_t*   h0b = (bf16_t*)take((size_t)2 * BATCH * HID * 2);        // 128 KB
  bf16_t*   h1b = (bf16_t*)take((size_t)2 * BATCH * HID * 2);        // 128 KB
  unsigned* bar = (unsigned*)take(256);

  // re-initialize mutable state every launch (graph replays don't re-poison)
  zero_words<<<64, 256, 0, stream>>>((unsigned*)h0b, 2 * BATCH * HID * 2 / 4);
  zero_words<<<64, 256, 0, stream>>>((unsigned*)h1b, 2 * BATCH * HID * 2 / 4);
  zero_words<<<1, 64, 0, stream>>>(bar, 64);

  pack_w0<<<512, 256, 0, stream>>>(W0, Wt0);
  pack_w1<<<512, 256, 0, stream>>>(W1, Wt1);
  embed_gather<<<512, 256, 0, stream>>>(x, emb, xe);

  lstm_persistent<<<NWG, NTHREADS, 0, stream>>>(xe, Wt0, Wt1, h0b, h1b, b0, b1, bar);

  // final h1 written at t=127 lands in buffer 0
  head_kernel<<<1, 64, 0, stream>>>(h1b, Wd, bd, Wpe, bpe, Wbur, bbur, out);
}
